// MultiBranchFusion_65317862638127
// MI455X (gfx1250) — matrix operand order repaired and checked by reading, not compiled, not run on hardware
//
#include <hip/hip_runtime.h>
#include <hip/hip_bf16.h>

// ---------------------------------------------------------------------------
// MultiBranchFusion on MI455X (gfx1250, wave32, WMMA bf16)
//
// out = relu( relu((X@W1c + b1) * (Y@W2c + b2)) @ W3c + sum_c b3 )
// N=131072, FST=256, SCD=384, J=CARD*SUB=384, REPR=384
//
// Strategy:
//  * prologue kernel: reshape weights to bf16, column-major W[j][k] in d_ws
//    (768 KB total -> lives in 192MB L2; B-fragments load directly from L2)
//  * main kernel: 1 WG = 256 thr = 8 waves computes a 64-row x 384-col tile.
//    X/Y staged to LDS as bf16; three chained GEMMs via
//    v_wmma_f32_16x16x32_bf16; 'a' spilled bf16 to LDS, fused in place with
//    'g', one barrier, then GEMM3 reads H from LDS as its A-matrix.
// ---------------------------------------------------------------------------

typedef __attribute__((ext_vector_type(16))) __bf16 v16bf;
typedef __attribute__((ext_vector_type(8)))  float  v8f;

#define FSTD 256
#define SCDD 384
#define REPRD 384
#define CARDD 16
#define SUBD 24
#define JD 384         // CARD*SUB
#define MT 64          // rows per workgroup
#define NWAVE 8
#define CPW 3          // column tiles (of 16) per wave: 8*3 = 24 tiles = 384 cols
#define RTN 4          // row tiles (MT/16)

// LDS layout (bytes)
#define XS_B 0
#define YS_B (MT*FSTD*2)                       // 32768
#define HS_B (YS_B + MT*SCDD*2)                // + 49152
#define B3_B (HS_B + MT*JD*2)                  // + 49152
#define LDS_BYTES (B3_B + REPRD*4)             // + 1536 = 132608

union ABfrag { v16bf v; uint4 q[2]; };

__global__ __launch_bounds__(256) void mbf_wconv(
    const float* __restrict__ W1, const float* __restrict__ W2,
    const float* __restrict__ W3,
    __bf16* __restrict__ w1b, __bf16* __restrict__ w2b,
    __bf16* __restrict__ w3b)
{
    int idx = blockIdx.x * 256 + threadIdx.x;
    if (idx < JD * FSTD) {                      // W1c column-major: w1b[j*256+k]
        int j = idx / FSTD, k = idx % FSTD;
        w1b[idx] = (__bf16)W1[(j / SUBD) * (FSTD * SUBD) + k * SUBD + (j % SUBD)];
    }
    if (idx < JD * SCDD) {                      // W2c column-major: w2b[j*384+k]
        int j = idx / SCDD, k = idx % SCDD;
        w2b[idx] = (__bf16)W2[(j / SUBD) * (SCDD * SUBD) + k * SUBD + (j % SUBD)];
        // W3c column-major over output r: w3b[r*384 + k], k = c*24+s
        int r = j;
        w3b[idx] = (__bf16)W3[(k / SUBD) * (SUBD * REPRD) + (k % SUBD) * REPRD + r];
    }
}

__global__ __launch_bounds__(256) void mbf_fused(
    const float* __restrict__ X, const float* __restrict__ Y,
    const float* __restrict__ b1, const float* __restrict__ b2,
    const float* __restrict__ b3,
    const __bf16* __restrict__ w1b, const __bf16* __restrict__ w2b,
    const __bf16* __restrict__ w3b,
    float* __restrict__ out)
{
    extern __shared__ char smem[];
    __bf16* xs  = (__bf16*)(smem + XS_B);   // [64][256] bf16
    __bf16* ys  = (__bf16*)(smem + YS_B);   // [64][384] bf16
    __bf16* hs  = (__bf16*)(smem + HS_B);   // [64][384] bf16 (a, then h in place)
    float*  b3s = (float*) (smem + B3_B);   // [384]

    const int tid  = threadIdx.x;
    const int wave = tid >> 5;
    const int lane = tid & 31;
    const int col  = lane & 15;             // A: row-in-tile, B/C: col-in-tile
    const int hi   = lane >> 4;
    const long n0  = (long)blockIdx.x * MT;

    // ---- stage X, Y (f32 -> bf16) into LDS; precompute sum_c b3 ----
    {
        const float* xg = X + n0 * FSTD;
        for (int i = tid; i < MT * FSTD; i += 256) xs[i] = (__bf16)xg[i];
        const float* yg = Y + n0 * SCDD;
        for (int i = tid; i < MT * SCDD; i += 256) ys[i] = (__bf16)yg[i];
        if (tid < REPRD) {
            float s = 0.f;
#pragma unroll
            for (int c = 0; c < CARDD; ++c) s += b3[c * REPRD + tid];
            b3s[tid] = s;
        }
    }
    __syncthreads();

    v8f acc[RTN][CPW];

    // =================== GEMM1: a = X @ W1c + b1 (K = 256) ===================
#pragma unroll
    for (int r = 0; r < RTN; ++r)
#pragma unroll
        for (int c = 0; c < CPW; ++c) acc[r][c] = (v8f){0,0,0,0,0,0,0,0};

    for (int k0 = 0; k0 < FSTD; k0 += 32) {
        ABfrag bf[CPW];
#pragma unroll
        for (int c = 0; c < CPW; ++c) {
            int j = (wave * CPW + c) * 16 + col;
            const __bf16* p = w1b + j * FSTD + k0 + hi * 16;   // 16 contiguous K
            bf[c].q[0] = *(const uint4*)p;
            bf[c].q[1] = *(const uint4*)(p + 8);
        }
#pragma unroll
        for (int r = 0; r < RTN; ++r) {
            ABfrag af;
            const __bf16* p = xs + (r * 16 + col) * FSTD + k0 + hi * 16;
            af.q[0] = *(const uint4*)p;          // K = k0+16*hi .. +7
            af.q[1] = *(const uint4*)(p + 8);    // K = k0+16*hi+8 .. +15
#pragma unroll
            for (int c = 0; c < CPW; ++c)
                acc[r][c] = __builtin_amdgcn_wmma_f32_16x16x32_bf16(
                    false, af.v, false, bf[c].v, (short)0, acc[r][c], false, false);
        }
    }
    // spill a (+b1) to LDS as bf16 (same-wave addresses; no barrier needed)
#pragma unroll
    for (int r = 0; r < RTN; ++r)
#pragma unroll
        for (int c = 0; c < CPW; ++c) {
            int j = (wave * CPW + c) * 16 + col;
            float bias = b1[j];
            float* av = (float*)&acc[r][c];
#pragma unroll
            for (int i = 0; i < 8; ++i)
                hs[(r * 16 + i + 8 * hi) * JD + j] = (__bf16)(av[i] + bias);
        }

    // =================== GEMM2: g = Y @ W2c + b2 (K = 384) ===================
#pragma unroll
    for (int r = 0; r < RTN; ++r)
#pragma unroll
        for (int c = 0; c < CPW; ++c) acc[r][c] = (v8f){0,0,0,0,0,0,0,0};

    for (int k0 = 0; k0 < SCDD; k0 += 32) {
        ABfrag bf[CPW];
#pragma unroll
        for (int c = 0; c < CPW; ++c) {
            int j = (wave * CPW + c) * 16 + col;
            const __bf16* p = w2b + j * SCDD + k0 + hi * 16;
            bf[c].q[0] = *(const uint4*)p;
            bf[c].q[1] = *(const uint4*)(p + 8);
        }
#pragma unroll
        for (int r = 0; r < RTN; ++r) {
            ABfrag af;
            const __bf16* p = ys + (r * 16 + col) * SCDD + k0 + hi * 16;
            af.q[0] = *(const uint4*)p;
            af.q[1] = *(const uint4*)(p + 8);
#pragma unroll
            for (int c = 0; c < CPW; ++c)
                acc[r][c] = __builtin_amdgcn_wmma_f32_16x16x32_bf16(
                    false, af.v, false, bf[c].v, (short)0, acc[r][c], false, false);
        }
    }
    // fuse in place: h = relu(a * g), bf16, same LDS addresses (same wave)
#pragma unroll
    for (int r = 0; r < RTN; ++r)
#pragma unroll
        for (int c = 0; c < CPW; ++c) {
            int j = (wave * CPW + c) * 16 + col;
            float bias = b2[j];
            float* gv = (float*)&acc[r][c];
#pragma unroll
            for (int i = 0; i < 8; ++i) {
                int addr = (r * 16 + i + 8 * hi) * JD + j;
                float a = (float)hs[addr];
                float h = a * (gv[i] + bias);
                hs[addr] = (__bf16)(h > 0.f ? h : 0.f);
            }
        }
    __syncthreads();   // GEMM3 reads all columns of h across waves

    // ============ GEMM3: out = relu(H @ W3c + sum_c b3) (K = 384) ============
#pragma unroll
    for (int r = 0; r < RTN; ++r)
#pragma unroll
        for (int c = 0; c < CPW; ++c) acc[r][c] = (v8f){0,0,0,0,0,0,0,0};

    for (int k0 = 0; k0 < JD; k0 += 32) {
        ABfrag bf[CPW];
#pragma unroll
        for (int c = 0; c < CPW; ++c) {
            int rr = (wave * CPW + c) * 16 + col;
            const __bf16* p = w3b + rr * JD + k0 + hi * 16;
            bf[c].q[0] = *(const uint4*)p;
            bf[c].q[1] = *(const uint4*)(p + 8);
        }
#pragma unroll
        for (int r = 0; r < RTN; ++r) {
            ABfrag af;
            const __bf16* p = hs + (r * 16 + col) * JD + k0 + hi * 16;
            af.q[0] = *(const uint4*)p;
            af.q[1] = *(const uint4*)(p + 8);
#pragma unroll
            for (int c = 0; c < CPW; ++c)
                acc[r][c] = __builtin_amdgcn_wmma_f32_16x16x32_bf16(
                    false, af.v, false, bf[c].v, (short)0, acc[r][c], false, false);
        }
    }
    // epilogue: + b3sum, relu, store f32
#pragma unroll
    for (int r = 0; r < RTN; ++r)
#pragma unroll
        for (int c = 0; c < CPW; ++c) {
            int rr = (wave * CPW + c) * 16 + col;
            float bias = b3s[rr];
            float* ov = (float*)&acc[r][c];
#pragma unroll
            for (int i = 0; i < 8; ++i) {
                float v = ov[i] + bias;
                out[(n0 + r * 16 + i + 8 * hi) * REPRD + rr] = v > 0.f ? v : 0.f;
            }
        }
}

extern "C" void kernel_launch(void* const* d_in, const int* in_sizes, int n_in,
                              void* d_out, int out_size, void* d_ws, size_t ws_size,
                              hipStream_t stream) {
    const float* X  = (const float*)d_in[0];
    const float* Y  = (const float*)d_in[1];
    const float* W1 = (const float*)d_in[2];
    const float* b1 = (const float*)d_in[3];
    const float* W2 = (const float*)d_in[4];
    const float* b2 = (const float*)d_in[5];
    const float* W3 = (const float*)d_in[6];
    const float* b3 = (const float*)d_in[7];
    float* out = (float*)d_out;

    __bf16* w1b = (__bf16*)d_ws;            // [384][256]
    __bf16* w2b = w1b + JD * FSTD;          // [384][384]
    __bf16* w3b = w2b + JD * SCDD;          // [384][384]

    int N = in_sizes[0] / FSTD;             // 131072
    int convBlocks = (JD * SCDD + 255) / 256;
    mbf_wconv<<<convBlocks, 256, 0, stream>>>(W1, W2, W3, w1b, w2b, w3b);
    mbf_fused<<<N / MT, 256, LDS_BYTES, stream>>>(X, Y, b1, b2, b3,
                                                  w1b, w2b, w3b, out);
}
